// LanguageGeneration_28518582845800
// MI455X (gfx1250) — compile-verified, matrix operand order checked by reading, not run on hardware
//
#include <hip/hip_runtime.h>

typedef __attribute__((ext_vector_type(16))) __bf16 v16bf;
typedef __attribute__((ext_vector_type(8)))  float  v8f;

#define BATCH 128
#define EMB   128
#define HID   512
#define VOCAB 32000
#define TMAX  30

// ---------- helpers ----------

__device__ __forceinline__ unsigned short f2bf_raw(float f) {
  unsigned u = __float_as_uint(f);
  unsigned r = u + 0x7FFFu + ((u >> 16) & 1u);   // round-to-nearest-even
  return (unsigned short)(r >> 16);
}

// Load one lane's 16x32 bf16 WMMA fragment from a precomputed row pointer
// (already offset by row*ld + 8*half). Per the ISA 16-bit A/B layout the lane
// needs K = [0..7] and [16..23] relative to that pointer: two b128 loads.
__device__ __forceinline__ v16bf load_frag(const unsigned short* __restrict__ p) {
  v16bf f;
  ((uint4*)&f)[0] = *(const uint4*)(p);
  ((uint4*)&f)[1] = *(const uint4*)(p + 16);
  return f;
}

__device__ __forceinline__ v8f wmma_bf16(v16bf a, v16bf b, v8f c) {
  return __builtin_amdgcn_wmma_f32_16x16x32_bf16(false, a, false, b,
                                                 (short)0, c, false, false);
}

__device__ __forceinline__ float sigm(float x) { return 1.0f / (1.0f + __expf(-x)); }

// ---------- one-time kernels ----------

__global__ void cvt_bf16_kernel(const float* __restrict__ src,
                                unsigned short* __restrict__ dst, int n) {
  int i = blockIdx.x * blockDim.x + threadIdx.x;
  int stride = gridDim.x * blockDim.x;
  for (; i < n; i += stride) dst[i] = f2bf_raw(src[i]);
}

__global__ void init_kernel(const float* __restrict__ hidden,
                            const int* __restrict__ sos,
                            unsigned short* __restrict__ h1bf,
                            unsigned short* __restrict__ h2bf,
                            float* __restrict__ c1, float* __restrict__ c2,
                            int* __restrict__ tok) {
  int i = blockIdx.x * blockDim.x + threadIdx.x;
  if (i < BATCH * HID) {
    h1bf[i] = f2bf_raw(hidden[i]);
    h2bf[i] = 0;          // bf16 +0.0
    c1[i] = 0.0f;
    c2[i] = 0.0f;
  }
  if (i < BATCH) tok[i] = sos[0];
}

// ---------- per-step kernels ----------

__global__ void embed_kernel(const float* __restrict__ emb,
                             const int* __restrict__ tok,
                             unsigned short* __restrict__ xbf) {
  int b = blockIdx.x;
  int k = threadIdx.x;                       // blockDim = EMB = 128
  xbf[b * EMB + k] = f2bf_raw(emb[(size_t)tok[b] * EMB + k]);
}

// Fused LSTM cell: gates = A0*B0^T + A1*B1^T (+biases), then cell update.
// Grid (8 mtiles, 8 col-chunks of 64), 128 threads = 4 waves = 4 gates.
// K-loop split into two phases so each inner loop is branch/select-free.
__global__ __launch_bounds__(128)
void lstm_kernel(const unsigned short* __restrict__ A0, int K0,
                 const unsigned short* __restrict__ A1, int K1,
                 const unsigned short* __restrict__ B0,   // [4H x K0]
                 const unsigned short* __restrict__ B1,   // [4H x K1]
                 const float* __restrict__ b_ih, const float* __restrict__ b_hh,
                 float* __restrict__ c_state,             // [B x H], in-place
                 unsigned short* __restrict__ h_out) {    // [B x H] bf16
  __shared__ float gbuf[4][16][64 + 1];
  const int gate = threadIdx.x >> 5;         // wave id = gate (i,f,g,o)
  const int lane = threadIdx.x & 31;
  const int half = lane >> 4;
  const int l16  = lane & 15;
  const int mbase   = blockIdx.x * 16;
  const int colbase = blockIdx.y * 64;
  const int arow = mbase + l16;

  // per-lane row base pointers (include the 8*half lane-layout offset)
  const unsigned short* a0p = A0 + (size_t)arow * K0 + 8 * half;
  const unsigned short* a1p = A1 + (size_t)arow * K1 + 8 * half;
  const unsigned short* b0p[4];
  const unsigned short* b1p[4];
#pragma unroll
  for (int nt = 0; nt < 4; ++nt) {
    const int nrow = gate * HID + colbase + nt * 16 + l16;
    b0p[nt] = B0 + (size_t)nrow * K0 + 8 * half;
    b1p[nt] = B1 + (size_t)nrow * K1 + 8 * half;
  }

  v8f acc[4] = {};
  for (int kk = 0; kk < K0; kk += 32) {
    v16bf afrag = load_frag(a0p + kk);
#pragma unroll
    for (int nt = 0; nt < 4; ++nt)
      acc[nt] = wmma_bf16(afrag, load_frag(b0p[nt] + kk), acc[nt]);
  }
#pragma unroll 2
  for (int kk = 0; kk < K1; kk += 32) {
    v16bf afrag = load_frag(a1p + kk);
#pragma unroll
    for (int nt = 0; nt < 4; ++nt)
      acc[nt] = wmma_bf16(afrag, load_frag(b1p[nt] + kk), acc[nt]);
  }

  // C/D layout: VGPR r -> row (r + 8*half), lane&15 -> column
#pragma unroll
  for (int nt = 0; nt < 4; ++nt)
#pragma unroll
    for (int r = 0; r < 8; ++r)
      gbuf[gate][r + 8 * half][nt * 16 + l16] = acc[nt][r];
  __syncthreads();

  // elementwise cell update: 16 rows x 64 cols
  for (int e = threadIdx.x; e < 16 * 64; e += 128) {
    const int rl = e >> 6, cl = e & 63;
    const int m = mbase + rl;
    const int j = colbase + cl;
    const float ig = gbuf[0][rl][cl] + b_ih[0 * HID + j] + b_hh[0 * HID + j];
    const float fg = gbuf[1][rl][cl] + b_ih[1 * HID + j] + b_hh[1 * HID + j];
    const float gg = gbuf[2][rl][cl] + b_ih[2 * HID + j] + b_hh[2 * HID + j];
    const float og = gbuf[3][rl][cl] + b_ih[3 * HID + j] + b_hh[3 * HID + j];
    const float cn = sigm(fg) * c_state[m * HID + j] + sigm(ig) * tanhf(gg);
    const float hn = sigm(og) * tanhf(cn);
    c_state[m * HID + j] = cn;
    h_out[m * HID + j] = f2bf_raw(hn);
  }
}

// logits[:, t, :] = h2 * Wproj^T + bias.  Grid (8 mtiles, 125 n-chunks of 256),
// 128 threads = 4 waves, each wave: 4 n-tiles sharing one A fragment.
__global__ __launch_bounds__(128)
void proj_kernel(const unsigned short* __restrict__ Abf,   // [B x HID]
                 const unsigned short* __restrict__ Wbf,   // [VOCAB x HID]
                 const float* __restrict__ bias,
                 float* __restrict__ out, int t) {
  const int wave = threadIdx.x >> 5;
  const int lane = threadIdx.x & 31;
  const int half = lane >> 4;
  const int l16  = lane & 15;
  const int mbase = blockIdx.x * 16;
  const int nbase = blockIdx.y * 256 + wave * 64;
  const int arow = mbase + l16;

  const unsigned short* ap = Abf + (size_t)arow * HID + 8 * half;
  const unsigned short* bp[4];
#pragma unroll
  for (int nt = 0; nt < 4; ++nt)
    bp[nt] = Wbf + (size_t)(nbase + nt * 16 + l16) * HID + 8 * half;

  v8f acc[4] = {};
#pragma unroll 2
  for (int kk = 0; kk < HID; kk += 32) {
    v16bf afrag = load_frag(ap + kk);
#pragma unroll
    for (int nt = 0; nt < 4; ++nt)
      acc[nt] = wmma_bf16(afrag, load_frag(bp[nt] + kk), acc[nt]);
  }

#pragma unroll
  for (int nt = 0; nt < 4; ++nt) {
    const int col = nbase + nt * 16 + l16;
    const float bb = bias[col];
#pragma unroll
    for (int r = 0; r < 8; ++r) {
      const int row = mbase + r + 8 * half;
      out[((size_t)row * TMAX + t) * VOCAB + col] = acc[nt][r] + bb;
    }
  }
}

// first-argmax over a logits row; one block per batch row
__global__ __launch_bounds__(256)
void argmax_kernel(const float* __restrict__ out, int t, int* __restrict__ tok) {
  const int b = blockIdx.x;
  const float* row = out + ((size_t)b * TMAX + t) * VOCAB;
  float best = -INFINITY;
  int bi = VOCAB;
  for (int v = threadIdx.x; v < VOCAB; v += 256) {
    const float val = row[v];
    if (val > best) { best = val; bi = v; }   // per-thread v strictly increases
  }
  __shared__ float sv[256];
  __shared__ int   si[256];
  sv[threadIdx.x] = best;
  si[threadIdx.x] = bi;
  __syncthreads();
  for (int s = 128; s > 0; s >>= 1) {
    if (threadIdx.x < s) {
      const float ov = sv[threadIdx.x + s];
      const int   oi = si[threadIdx.x + s];
      if (ov > sv[threadIdx.x] ||
          (ov == sv[threadIdx.x] && oi < si[threadIdx.x])) {
        sv[threadIdx.x] = ov;
        si[threadIdx.x] = oi;
      }
    }
    __syncthreads();
  }
  if (threadIdx.x == 0) tok[b] = si[0];
}

// ---------- host ----------

extern "C" void kernel_launch(void* const* d_in, const int* in_sizes, int n_in,
                              void* d_out, int out_size, void* d_ws, size_t ws_size,
                              hipStream_t stream) {
  (void)in_sizes; (void)n_in; (void)out_size; (void)ws_size;

  const float* hidden = (const float*)d_in[0];
  const float* emb    = (const float*)d_in[1];
  const float* W_ih1  = (const float*)d_in[2];
  const float* W_hh1  = (const float*)d_in[3];
  const float* b_ih1  = (const float*)d_in[4];
  const float* b_hh1  = (const float*)d_in[5];
  const float* W_ih2  = (const float*)d_in[6];
  const float* W_hh2  = (const float*)d_in[7];
  const float* b_ih2  = (const float*)d_in[8];
  const float* b_hh2  = (const float*)d_in[9];
  const float* W_proj = (const float*)d_in[10];
  const float* b_proj = (const float*)d_in[11];
  const int*   sos    = (const int*)d_in[13];
  float* out = (float*)d_out;

  // workspace carve-up (~39 MB)
  char* p = (char*)d_ws;
  auto carve = [&](size_t bytes) {
    char* q = p;
    p += (bytes + 255) & ~(size_t)255;
    return q;
  };
  unsigned short* Wih1b = (unsigned short*)carve((size_t)4 * HID * EMB * 2);
  unsigned short* Whh1b = (unsigned short*)carve((size_t)4 * HID * HID * 2);
  unsigned short* Wih2b = (unsigned short*)carve((size_t)4 * HID * HID * 2);
  unsigned short* Whh2b = (unsigned short*)carve((size_t)4 * HID * HID * 2);
  unsigned short* Wprjb = (unsigned short*)carve((size_t)VOCAB * HID * 2);
  unsigned short* xbf   = (unsigned short*)carve((size_t)BATCH * EMB * 2);
  unsigned short* h1a   = (unsigned short*)carve((size_t)BATCH * HID * 2);
  unsigned short* h1b   = (unsigned short*)carve((size_t)BATCH * HID * 2);
  unsigned short* h2a   = (unsigned short*)carve((size_t)BATCH * HID * 2);
  unsigned short* h2b   = (unsigned short*)carve((size_t)BATCH * HID * 2);
  float* c1 = (float*)carve((size_t)BATCH * HID * 4);
  float* c2 = (float*)carve((size_t)BATCH * HID * 4);
  int* tok  = (int*)carve((size_t)BATCH * 4);

  auto cvt = [&](const float* s, unsigned short* d, int n) {
    int grid = (n + 255) / 256;
    if (grid > 4096) grid = 4096;
    cvt_bf16_kernel<<<grid, 256, 0, stream>>>(s, d, n);
  };
  cvt(W_ih1, Wih1b, 4 * HID * EMB);
  cvt(W_hh1, Whh1b, 4 * HID * HID);
  cvt(W_ih2, Wih2b, 4 * HID * HID);
  cvt(W_hh2, Whh2b, 4 * HID * HID);
  cvt(W_proj, Wprjb, VOCAB * HID);

  init_kernel<<<(BATCH * HID + 255) / 256, 256, 0, stream>>>(hidden, sos, h1a, h2a,
                                                             c1, c2, tok);

  unsigned short* h1cur = h1a; unsigned short* h1nxt = h1b;
  unsigned short* h2cur = h2a; unsigned short* h2nxt = h2b;

  for (int t = 0; t < TMAX; ++t) {
    embed_kernel<<<BATCH, EMB, 0, stream>>>(emb, tok, xbf);
    lstm_kernel<<<dim3(BATCH / 16, HID / 64), 128, 0, stream>>>(
        xbf, EMB, h1cur, HID, Wih1b, Whh1b, b_ih1, b_hh1, c1, h1nxt);
    lstm_kernel<<<dim3(BATCH / 16, HID / 64), 128, 0, stream>>>(
        h1nxt, HID, h2cur, HID, Wih2b, Whh2b, b_ih2, b_hh2, c2, h2nxt);
    proj_kernel<<<dim3(BATCH / 16, VOCAB / 256), 128, 0, stream>>>(
        h2nxt, Wprjb, b_proj, out, t);
    argmax_kernel<<<BATCH, 256, 0, stream>>>(out, t, tok);
    unsigned short* tmp;
    tmp = h1cur; h1cur = h1nxt; h1nxt = tmp;
    tmp = h2cur; h2cur = h2nxt; h2nxt = tmp;
  }
}
